// MultiHeadAttention_40578851012981
// MI455X (gfx1250) — matrix operand order repaired and checked by reading, not compiled, not run on hardware
//
#include <hip/hip_runtime.h>
#include <hip/hip_bf16.h>

// MHA forward for MI455X (gfx1250, wave32).
// B=2, S=2048, H=1024, N_HEAD=16, d=64 (fixed per reference setup_inputs).
// All GEMM-shaped math runs on v_wmma_f32_16x16x32_f16 (f16 in, f32 acc).
// CDNA5 paths: v_wmma, global_load_async_to_lds_b128 (+s_wait_asynccnt),
// global_prefetch_b8.
// Workspace layout (f16): Q | K | V | attn_out, 4 x 4194304 halves = 32 MB.

typedef __attribute__((ext_vector_type(16))) _Float16 v16h;
typedef __attribute__((ext_vector_type(2)))  _Float16 v2h;
typedef __attribute__((ext_vector_type(8)))  float    v8f;

#define S_LEN   2048
#define HID     1024
#define NHEAD   16
#define DHEAD   64
#define MROWS   4096   // B*S

__device__ __forceinline__ v8f wmma16(v16h a, v16h b, v8f c) {
  return __builtin_amdgcn_wmma_f32_16x16x32_f16(
      /*neg_a=*/false, a, /*neg_b=*/false, b,
      /*c_mod=*/(short)0, c, /*reuse_a=*/false, /*reuse_b=*/false);
}

// A-matrix 16x32 f16 fragment (ISA 7.12.2): lane m = lane&15, hi = lane>>4,
// VGPR p holds K = (p>>2)*16 + (p&3)*2 + hi*8 + {0,1}. Adjacent halves are
// K-contiguous -> b32 per VGPR (compiler fuses to ds_load_b128).
__device__ __forceinline__ v16h load_a16(const _Float16* base, int rowStride, int k0) {
  int lane = threadIdx.x & 31;
  int m = lane & 15, hi = lane >> 4;
  union { v16h h; unsigned u[8]; } f;
  const _Float16* row = base + (size_t)m * rowStride + k0 + (hi << 3);
#pragma unroll
  for (int p = 0; p < 8; ++p) {
    int k = ((p >> 2) << 4) + ((p & 3) << 1);
    f.u[p] = *(const unsigned*)(row + k);
  }
  return f.h;
}

// B-matrix 32x16 f16 fragment: lane n = lane&15, hi = lane>>4,
// VGPR p holds K = (p>>2)*16 + (p&3)*2 + hi*8 + {0,1}. Source laid out [n][k] (K-contiguous).
__device__ __forceinline__ v16h load_b16(const _Float16* base, int rowStride,
                                         int nBase, int k0) {
  int lane = threadIdx.x & 31;
  int n = lane & 15, hi = lane >> 4;
  union { v16h h; unsigned u[8]; } f;
  const _Float16* row = base + (size_t)(nBase + n) * rowStride + k0 + (hi << 3);
#pragma unroll
  for (int p = 0; p < 8; ++p) f.u[p] = *(const unsigned*)(row + ((p >> 2) << 4) + ((p & 3) << 1));
  return f.h;
}

// Butterfly reductions across the 16 lanes sharing one C-matrix row.
__device__ __forceinline__ float rowmax16(float v) {
#pragma unroll
  for (int m = 1; m <= 8; m <<= 1) v = fmaxf(v, __shfl_xor(v, m, 32));
  return v;
}
__device__ __forceinline__ float rowsum16(float v) {
#pragma unroll
  for (int m = 1; m <= 8; m <<= 1) v += __shfl_xor(v, m, 32);
  return v;
}

// f32x4 -> f16x4; vector construction pairs the converts into
// v_cvt_pk_f16_f32 without repack moves (verified in round-4 asm).
__device__ __forceinline__ unsigned long long cvt4(float4 f) {
  union { v2h h; unsigned u; } a, b;
  a.h = (v2h){(_Float16)f.x, (_Float16)f.y};
  b.h = (v2h){(_Float16)f.z, (_Float16)f.w};
  return ((unsigned long long)b.u << 32) | (unsigned long long)a.u;
}

// ---------------------------------------------------------------------------
// Kernel 1: q/k/v = x @ W^T (+RoPE for q,k). Block = 256 thr (8 waves),
// tile 128(M) x 64(N), K-step 32. blockIdx.z picks the weight.
// Output f16 in [B][NHEAD][S][DHEAD] layout.
// ---------------------------------------------------------------------------
__global__ __launch_bounds__(256) void mha_qkv_rope_kernel(
    const float* __restrict__ x,
    const float* __restrict__ wq, const float* __restrict__ wk,
    const float* __restrict__ wv,
    _Float16* __restrict__ Q, _Float16* __restrict__ K,
    _Float16* __restrict__ V) {
  __shared__ _Float16 ldsX[128 * 32];
  __shared__ _Float16 ldsW[64 * 32];

  int tid = threadIdx.x;
  int wave = tid >> 5, lane = tid & 31;
  int lo = lane & 15, hi = lane >> 4;
  int mBlock = blockIdx.x * 128;
  int nBase  = blockIdx.y * 64;
  int which  = blockIdx.z;                // 0=q, 1=k, 2=v
  const float* w = (which == 0) ? wq : (which == 1) ? wk : wv;
  _Float16*  dst = (which == 0) ? Q  : (which == 1) ? K  : V;

  v8f acc[4];
#pragma unroll
  for (int j = 0; j < 4; ++j) acc[j] = (v8f){0.f,0.f,0.f,0.f,0.f,0.f,0.f,0.f};

  for (int kb = 0; kb < HID; kb += 32) {
    // Issue ALL tile loads first so they pipeline (one wait, not six).
    float4 fx[4], fw[2];
#pragma unroll
    for (int i = 0; i < 4; ++i) {
      int t = tid + i * 256;            // float4 index, 8 per row
      int r = t >> 3, c4 = t & 7;
      fx[i] = *(const float4*)(x + (size_t)(mBlock + r) * HID + kb + c4 * 4);
    }
#pragma unroll
    for (int i = 0; i < 2; ++i) {
      int t = tid + i * 256;
      int r = t >> 3, c4 = t & 7;
      fw[i] = *(const float4*)(w + (size_t)(nBase + r) * HID + kb + c4 * 4);
    }
    // Convert + stage to LDS (packed f16, 8B stores).
#pragma unroll
    for (int i = 0; i < 4; ++i) {
      int t = tid + i * 256;
      int r = t >> 3, c4 = t & 7;
      *(unsigned long long*)&ldsX[r * 32 + c4 * 4] = cvt4(fx[i]);
    }
#pragma unroll
    for (int i = 0; i < 2; ++i) {
      int t = tid + i * 256;
      int r = t >> 3, c4 = t & 7;
      *(unsigned long long*)&ldsW[r * 32 + c4 * 4] = cvt4(fw[i]);
    }
    if (kb + 32 < HID) {                // hint next tile into cache
      __builtin_prefetch(x + (size_t)(mBlock + (tid >> 1)) * HID + kb + 32, 0, 0);
      __builtin_prefetch(w + (size_t)(nBase + (tid >> 2)) * HID + kb + 32, 0, 0);
    }
    __syncthreads();

    v16h a = load_a16(ldsX + wave * 16 * 32, 32, 0);
#pragma unroll
    for (int j = 0; j < 4; ++j)
      acc[j] = wmma16(a, load_b16(ldsW, 32, j * 16, 0), acc[j]);
    __syncthreads();
  }

  // Epilogue: RoPE (q,k only) + scatter f16 to [B][H][S][d].
  int mWave = mBlock + wave * 16;
  const float LOG1E4_OVER_D = 9.210340371976184f / (float)DHEAD; // ln(10000)/64
#pragma unroll
  for (int j = 0; j < 4; ++j) {
#pragma unroll
    for (int r = 0; r < 8; ++r) {
      float v = acc[j][r];
      int m = mWave + r + 8 * hi;            // global row in [0, B*S)
      int n = nBase + j * 16 + lo;           // global col in [0, H)
      int dd = n & (DHEAD - 1);
      float o = v;
      float partner = __shfl_xor(v, 1, 32);  // lane^1 holds the paired dim
      if (which < 2) {
        int s = m & (S_LEN - 1);
        float inv = __expf(-(float)(dd & ~1) * LOG1E4_OVER_D); // 10000^(-2i/d)
        float ang = (float)s * inv;
        float c = __cosf(ang), sn = __sinf(ang);
        o = (dd & 1) ? (partner * sn + v * c) : (v * c - partner * sn);
      }
      int b  = m >> 11;                       // / S_LEN
      int s_ = m & (S_LEN - 1);
      int h  = n >> 6;                        // / DHEAD
      dst[(((size_t)b * NHEAD + h) * S_LEN + s_) * DHEAD + dd] = (_Float16)o;
    }
  }
}

// ---------------------------------------------------------------------------
// Kernel 2: attention. One wave per (b, h, 16-row q tile). Streams 32-key
// tiles. The K tile (contiguous 4KB) is brought in with the CDNA5 async
// DMA path: global_load_async_to_lds_b128 + s_wait_asynccnt.
// NOTE reference semantics: softmax denominator over ALL keys (unmasked),
// causal zeroing applied only to the O accumulation.
// ---------------------------------------------------------------------------
__global__ __launch_bounds__(32) void mha_attn_kernel(
    const _Float16* __restrict__ Q, const _Float16* __restrict__ K,
    const _Float16* __restrict__ V, _Float16* __restrict__ O) {
  __shared__ _Float16 ldsK[32 * 64];   // [key][dd]   -> B frag for Q.K^T
  __shared__ _Float16 ldsVt[64 * 32];  // [dd][key]   -> B frag for P.V
  __shared__ _Float16 ldsP[16 * 32];   // masked probabilities -> A frag

  int lane = threadIdx.x;
  int lo = lane & 15, hi = lane >> 4;
  int qbase = blockIdx.x * 16;
  int h = blockIdx.y, b = blockIdx.z;
  size_t headBase = ((size_t)b * NHEAD + h) * (size_t)S_LEN * DHEAD;

  // LDS aperture flat address carries the LDS byte offset in its low 32 bits
  // (ISA 10.2).
  unsigned ldsKoff = (unsigned)(unsigned long long)(const void*)ldsK;

  // Q fragments for this 16-row tile, d split into two K=32 chunks.
  const _Float16* qp = Q + headBase + (size_t)qbase * DHEAD;
  v16h a0 = load_a16(qp, DHEAD, 0);
  v16h a1 = load_a16(qp, DHEAD, 32);

  v8f accO[4];
#pragma unroll
  for (int j = 0; j < 4; ++j) accO[j] = (v8f){0.f,0.f,0.f,0.f,0.f,0.f,0.f,0.f};
  float mrow[8], lrow[8];
#pragma unroll
  for (int r = 0; r < 8; ++r) { mrow[r] = -1e30f; lrow[r] = 0.f; }

  const v8f vzero = (v8f){0.f,0.f,0.f,0.f,0.f,0.f,0.f,0.f};

  for (int kt = 0; kt < S_LEN; kt += 32) {
    // K tile: 32x64 halves = 4KB contiguous -> async DMA straight to LDS,
    // 16B per lane per issue, 8 issues (ASYNCcnt-tracked, no VGPR transit).
    unsigned long long kgBase =
        (unsigned long long)(K + headBase + (size_t)kt * DHEAD);
#pragma unroll
    for (int i = 0; i < 8; ++i) {
      unsigned       loff = ldsKoff + (unsigned)(lane + i * 32) * 16u;
      unsigned long long ga = kgBase + (unsigned long long)(lane + i * 32) * 16u;
      asm volatile("global_load_async_to_lds_b128 %0, %1, off"
                   :: "v"(loff), "v"(ga) : "memory");
    }
    // V tile transposed (16-bit scatter): thread `lane` owns key row kt+lane.
    const _Float16* vg = V + headBase + (size_t)(kt + lane) * DHEAD;
#pragma unroll
    for (int c = 0; c < 64; ++c) ldsVt[c * 32 + lane] = vg[c];
    asm volatile("s_wait_asynccnt 0x0" ::: "memory");
    __syncthreads();

    // scores[16 x 32] = Q(16x64) . K^T, two 16-wide key chunks, /sqrt(d).
    v8f s0 = wmma16(a1, load_b16(ldsK, 64, 0, 32),
                    wmma16(a0, load_b16(ldsK, 64, 0, 0), vzero));
    v8f s1 = wmma16(a1, load_b16(ldsK, 64, 16, 32),
                    wmma16(a0, load_b16(ldsK, 64, 16, 0), vzero));

#pragma unroll
    for (int r = 0; r < 8; ++r) {
      float e0 = s0[r] * 0.125f;           // 1/sqrt(64)
      float e1 = s1[r] * 0.125f;
      float rm = rowmax16(fmaxf(e0, e1));
      float nm = fmaxf(mrow[r], rm);
      float corr = __expf(mrow[r] - nm);
      float p0 = __expf(e0 - nm);
      float p1 = __expf(e1 - nm);
      lrow[r] = lrow[r] * corr + rowsum16(p0 + p1);   // UNMASKED denominator
      mrow[r] = nm;
#pragma unroll
      for (int j = 0; j < 4; ++j) accO[j][r] = accO[j][r] * corr;
      // Causal mask only for the numerator (post-softmax zeroing).
      int qrow = qbase + r + 8 * hi;
      ldsP[(r + 8 * hi) * 32 + lo]      = (_Float16)((kt + lo)      <= qrow ? p0 : 0.f);
      ldsP[(r + 8 * hi) * 32 + 16 + lo] = (_Float16)((kt + 16 + lo) <= qrow ? p1 : 0.f);
    }
    __syncthreads();

    v16h ap = load_a16(ldsP, 32, 0);
#pragma unroll
    for (int j = 0; j < 4; ++j)
      accO[j] = wmma16(ap, load_b16(ldsVt, 32, j * 16, 0), accO[j]);
    __syncthreads();
  }

  // O /= l, store f16 to [B][S][H*DHEAD] (row-major for the output GEMM).
#pragma unroll
  for (int j = 0; j < 4; ++j) {
#pragma unroll
    for (int r = 0; r < 8; ++r) {
      float o = accO[j][r] / lrow[r];
      int s_ = qbase + r + 8 * hi;
      O[((size_t)b * S_LEN + s_) * HID + h * DHEAD + j * 16 + lo] = (_Float16)o;
    }
  }
}

// ---------------------------------------------------------------------------
// Kernel 3: out = attn(f16) @ w_o^T, f32 output. Same tiling as kernel 1.
// ---------------------------------------------------------------------------
__global__ __launch_bounds__(256) void mha_out_proj_kernel(
    const _Float16* __restrict__ A, const float* __restrict__ Wo,
    float* __restrict__ out) {
  __shared__ _Float16 ldsX[128 * 32];
  __shared__ _Float16 ldsW[64 * 32];

  int tid = threadIdx.x;
  int wave = tid >> 5, lane = tid & 31;
  int lo = lane & 15, hi = lane >> 4;
  int mBlock = blockIdx.x * 128;
  int nBase  = blockIdx.y * 64;

  v8f acc[4];
#pragma unroll
  for (int j = 0; j < 4; ++j) acc[j] = (v8f){0.f,0.f,0.f,0.f,0.f,0.f,0.f,0.f};

  for (int kb = 0; kb < HID; kb += 32) {
    // Issue ALL tile loads first, then convert/store (single wait).
    uint4  ua[2];
    float4 fw[2];
#pragma unroll
    for (int i = 0; i < 2; ++i) {
      int t = tid + i * 256;            // uint4 index, 4 per row
      int r = t >> 2, c8 = t & 3;
      ua[i] = *(const uint4*)(A + (size_t)(mBlock + r) * HID + kb + c8 * 8);
    }
#pragma unroll
    for (int i = 0; i < 2; ++i) {
      int t = tid + i * 256;
      int r = t >> 3, c4 = t & 7;
      fw[i] = *(const float4*)(Wo + (size_t)(nBase + r) * HID + kb + c4 * 4);
    }
#pragma unroll
    for (int i = 0; i < 2; ++i) {
      int t = tid + i * 256;
      int r = t >> 2, c8 = t & 3;
      *(uint4*)&ldsX[r * 32 + c8 * 8] = ua[i];
    }
#pragma unroll
    for (int i = 0; i < 2; ++i) {
      int t = tid + i * 256;
      int r = t >> 3, c4 = t & 7;
      *(unsigned long long*)&ldsW[r * 32 + c4 * 4] = cvt4(fw[i]);
    }
    if (kb + 32 < HID)
      __builtin_prefetch(A + (size_t)(mBlock + (tid >> 1)) * HID + kb + 32, 0, 0);
    __syncthreads();

    v16h a = load_a16(ldsX + wave * 16 * 32, 32, 0);
#pragma unroll
    for (int j = 0; j < 4; ++j)
      acc[j] = wmma16(a, load_b16(ldsW, 32, j * 16, 0), acc[j]);
    __syncthreads();
  }

  int mWave = mBlock + wave * 16;
#pragma unroll
  for (int j = 0; j < 4; ++j) {
#pragma unroll
    for (int r = 0; r < 8; ++r) {
      int m = mWave + r + 8 * hi;
      int n = nBase + j * 16 + lo;
      out[(size_t)m * HID + n] = acc[j][r];
    }
  }
}

extern "C" void kernel_launch(void* const* d_in, const int* in_sizes, int n_in,
                              void* d_out, int out_size, void* d_ws, size_t ws_size,
                              hipStream_t stream) {
  const float* x  = (const float*)d_in[0];
  const float* wq = (const float*)d_in[1];
  const float* wk = (const float*)d_in[2];
  const float* wv = (const float*)d_in[3];
  const float* wo = (const float*)d_in[4];
  float* out = (float*)d_out;

  // Workspace: Q | K | V | attn, each B*S*H = 4194304 f16 (32 MB total).
  const size_t ELEMS = (size_t)MROWS * HID;
  _Float16* Q    = (_Float16*)d_ws;
  _Float16* K    = Q + ELEMS;
  _Float16* V    = K + ELEMS;
  _Float16* Attn = V + ELEMS;
  (void)in_sizes; (void)n_in; (void)out_size; (void)ws_size;

  // 1) QKV projections + RoPE.  grid: (M/128, H/64, 3 weights)
  mha_qkv_rope_kernel<<<dim3(MROWS / 128, HID / 64, 3), 256, 0, stream>>>(
      x, wq, wk, wv, Q, K, V);

  // 2) Attention (flash-style, reference's post-softmax causal zeroing).
  mha_attn_kernel<<<dim3(S_LEN / 16, NHEAD, 2), 32, 0, stream>>>(Q, K, V, Attn);

  // 3) Output projection.
  mha_out_proj_kernel<<<dim3(MROWS / 128, HID / 64), 256, 0, stream>>>(
      Attn, wo, out);
}
